// SceneReasoningModule_90598040141847
// MI455X (gfx1250) — matrix operand order repaired and checked by reading, not compiled, not run on hardware
//
#include <hip/hip_runtime.h>
#include <math.h>

typedef __attribute__((ext_vector_type(16))) __bf16 bf16x16;
typedef __attribute__((ext_vector_type(8)))  float  v8f;
typedef unsigned short u16;
typedef unsigned int   u32;

union Frag { bf16x16 v; u32 u[8]; u16 h[16]; };

__device__ __forceinline__ u16 f2bf(float f) {
  union { float f; u32 u; } a; a.f = f;
  u32 r = (a.u + 0x7FFFu + ((a.u >> 16) & 1u)) >> 16;   // round-to-nearest-even
  return (u16)r;
}

// ---------------------------------------------------------------------------
// fp32 -> bf16 tensor convert (weights)
// ---------------------------------------------------------------------------
__global__ __launch_bounds__(256) void cvt_kernel(const float* __restrict__ s,
                                                  u16* __restrict__ d, int n) {
  int i = blockIdx.x * 256 + threadIdx.x;
  if (i < n) d[i] = f2bf(s[i]);
}

// ---------------------------------------------------------------------------
// pe[n, 0:64] @ pos_w[64, 768] + pos_b  ->  peproj[576, 768]
// ---------------------------------------------------------------------------
__global__ __launch_bounds__(256) void peproj_kernel(
    const float* __restrict__ rowe, const float* __restrict__ cole,
    const float* __restrict__ pw, const float* __restrict__ pb,
    float* __restrict__ out) {
  __shared__ float spe[64];
  const int tid = threadIdx.x;
  int n = blockIdx.x, r = n / 24, c = n % 24;
  if (tid < 32) spe[tid] = rowe[r * 32 + tid];
  else if (tid < 64) spe[tid] = cole[c * 32 + (tid - 32)];
  __syncthreads();
  for (int j = 0; j < 3; j++) {
    int d = tid + j * 256;
    float acc = pb[d];
    #pragma unroll 8
    for (int s = 0; s < 64; s++) acc += spe[s] * pw[s * 768 + d];
    out[n * 768 + d] = acc;
  }
}

// xbuf = x + peproj (broadcast over batch)
__global__ __launch_bounds__(256) void addpe_kernel(const float* __restrict__ x,
                                                    const float* __restrict__ pep,
                                                    float* __restrict__ xo) {
  size_t i = (size_t)blockIdx.x * 256 + threadIdx.x;
  xo[i] = x[i] + pep[i % (576u * 768u)];
}

// relative-position bias table gather -> bias[576,576]
__global__ __launch_bounds__(256) void bias_kernel(const float* __restrict__ relb,
                                                   float* __restrict__ out) {
  int i = blockIdx.x * 256 + threadIdx.x;
  int ii = i / 576, jj = i % 576;
  int ri = ii / 24, ci = ii % 24, rj = jj / 24, cj = jj % 24;
  int idx = (ri - rj + 23) * 47 + (ci - cj + 23);
  out[i] = relb[idx];
}

// ---------------------------------------------------------------------------
// LayerNorm over D=768, one block (256 thr) per row, bf16 output
// ---------------------------------------------------------------------------
__global__ __launch_bounds__(256) void ln_kernel(
    const float* __restrict__ x, const float* __restrict__ g,
    const float* __restrict__ b, u16* __restrict__ out) {
  __shared__ float sb[256];
  const int tid = threadIdx.x;
  size_t base = (size_t)blockIdx.x * 768;
  float v0 = x[base + tid], v1 = x[base + tid + 256], v2 = x[base + tid + 512];
  sb[tid] = v0 + v1 + v2;
  __syncthreads();
  for (int off = 128; off > 0; off >>= 1) { if (tid < off) sb[tid] += sb[tid + off]; __syncthreads(); }
  float mean = sb[0] * (1.0f / 768.0f);
  __syncthreads();
  float d0 = v0 - mean, d1 = v1 - mean, d2 = v2 - mean;
  sb[tid] = d0 * d0 + d1 * d1 + d2 * d2;
  __syncthreads();
  for (int off = 128; off > 0; off >>= 1) { if (tid < off) sb[tid] += sb[tid + off]; __syncthreads(); }
  float r = rsqrtf(sb[0] * (1.0f / 768.0f) + 1e-5f);
  out[base + tid]       = f2bf(d0 * r * g[tid]       + b[tid]);
  out[base + tid + 256] = f2bf(d1 * r * g[tid + 256] + b[tid + 256]);
  out[base + tid + 512] = f2bf(d2 * r * g[tid + 512] + b[tid + 512]);
}

// ---------------------------------------------------------------------------
// bf16 WMMA GEMM: C[M,Nc] = A[M,K] (bf16 rowmajor) * Bw[K,Nc] (bf16 rowmajor)
// + bias[Nc].  Block tile 128x64, 8 waves (4x2), wave tile 32x32 (2x2 WMMA).
// mode 0: store bf16; mode 1: exact GELU then bf16; mode 2: outF[idx] += val.
// M,K,Nc assumed multiples of 128/32/64 (true here: M=18432, K in {768,3072}).
// ---------------------------------------------------------------------------
__global__ __launch_bounds__(256) void gemm_bf16_kernel(
    const u16* __restrict__ A, const u16* __restrict__ Bw,
    const float* __restrict__ bias, int K, int Nc, int mode,
    float* outF, u16* outB) {
  __shared__ u16 Asub[128 * 32];
  __shared__ u16 Bsub[64 * 32];
  const int tid = threadIdx.x;
  const int lane = tid & 31, wave = tid >> 5;
  const int waveM = wave >> 1, waveN = wave & 1;
  const int mBase = blockIdx.x * 128;
  const int nBase = blockIdx.y * 64;
  const int aRow = tid >> 1, aSeg = tid & 1;   // A staging: 128 rows x 32 cols
  const int bRow = tid >> 3, bSeg = tid & 7;   // B staging: 32 rows x 64 cols
  const int m = lane & 15, hi = lane >> 4;

  v8f acc[2][2];
  #pragma unroll
  for (int i = 0; i < 2; i++)
    #pragma unroll
    for (int j = 0; j < 2; j++)
      #pragma unroll
      for (int r = 0; r < 8; r++) acc[i][j][r] = 0.f;

  for (int k0 = 0; k0 < K; k0 += 32) {
    { // stage A tile (row-major, contiguous 64B per row)
      const uint4* src = reinterpret_cast<const uint4*>(
          A + (size_t)(mBase + aRow) * K + k0 + aSeg * 16);
      uint4 d0 = src[0], d1 = src[1];
      uint4* dst = reinterpret_cast<uint4*>(&Asub[aRow * 32 + aSeg * 16]);
      dst[0] = d0; dst[1] = d1;
    }
    { // stage B tile transposed: Bsub[n][k]
      union { uint4 q; u16 h[8]; } t;
      t.q = *reinterpret_cast<const uint4*>(
          Bw + (size_t)(k0 + bRow) * Nc + nBase + bSeg * 8);
      #pragma unroll
      for (int i = 0; i < 8; i++) Bsub[(bSeg * 8 + i) * 32 + bRow] = t.h[i];
    }
    __syncthreads();

    Frag a[2], b[2];
    #pragma unroll
    for (int v = 0; v < 8; v++) {
      int kk = ((v < 4) ? 2 * v : 8 + 2 * v) + 8 * hi;       // A-frag K pair
      a[0].u[v] = *reinterpret_cast<const u32*>(&Asub[(waveM * 32 + m) * 32 + kk]);
      a[1].u[v] = *reinterpret_cast<const u32*>(&Asub[(waveM * 32 + 16 + m) * 32 + kk]);
      int kb = ((v < 4) ? 2 * v : 8 + 2 * v) + 8 * hi;       // B-frag K pair
      b[0].u[v] = *reinterpret_cast<const u32*>(&Bsub[(waveN * 32 + m) * 32 + kb]);
      b[1].u[v] = *reinterpret_cast<const u32*>(&Bsub[(waveN * 32 + 16 + m) * 32 + kb]);
    }
    #pragma unroll
    for (int i = 0; i < 2; i++)
      #pragma unroll
      for (int j = 0; j < 2; j++)
        acc[i][j] = __builtin_amdgcn_wmma_f32_16x16x32_bf16(
            false, a[i].v, false, b[j].v, (short)0, acc[i][j], false, false);
    __syncthreads();
  }

  #pragma unroll
  for (int i = 0; i < 2; i++) {
    #pragma unroll
    for (int j = 0; j < 2; j++) {
      int gcol = nBase + waveN * 32 + j * 16 + m;
      float bv = bias[gcol];
      #pragma unroll
      for (int r = 0; r < 8; r++) {
        int grow = mBase + waveM * 32 + i * 16 + 8 * hi + r;
        size_t idx = (size_t)grow * Nc + gcol;
        float val = acc[i][j][r] + bv;
        if (mode == 0) {
          outB[idx] = f2bf(val);
        } else if (mode == 1) {
          val = 0.5f * val * (1.0f + erff(val * 0.70710678118654752f));
          outB[idx] = f2bf(val);
        } else {
          outF[idx] = outF[idx] + val;                       // fused residual
        }
      }
    }
  }
}

// ---------------------------------------------------------------------------
// Attention: one block per (b, h, 16-row tile).  S = QK^T*scale + bias (WMMA),
// block softmax in LDS, O = P V (WMMA, f32->bf16 frag packing from LDS),
// cross-wave reduce with ds_add_f32 atomics.  Outputs bf16 [B,N,H*HD].
// ---------------------------------------------------------------------------
#define NSEQ 576
__global__ __launch_bounds__(256) void attn_kernel(
    const u16* __restrict__ q, const u16* __restrict__ kx,
    const u16* __restrict__ vx, const float* __restrict__ biasNN,
    u16* __restrict__ outB) {
  __shared__ float S[16 * NSEQ];
  __shared__ float O[16 * 96];
  __shared__ float red[256];

  const int tid = threadIdx.x;
  const int lane = tid & 31, wave = tid >> 5;
  const int rt = blockIdx.x, h = blockIdx.y, bb = blockIdx.z;
  const int m = lane & 15, hi = lane >> 4;
  const size_t seqBase = (size_t)bb * NSEQ;
  const int rowBase = rt * 16;
  const float scale = 0.1020620726159658f;  // 96^-0.5

  for (int i = tid; i < 16 * 96; i += 256) O[i] = 0.f;

  // Q fragments (16 rows x 96 K), reused for all column tiles
  Frag qf[3];
  {
    size_t rowOff = (seqBase + rowBase + m) * 768 + h * 96;
    #pragma unroll
    for (int c = 0; c < 3; c++)
      #pragma unroll
      for (int v = 0; v < 8; v++) {
        int kk = c * 32 + ((v < 4) ? 2 * v : 8 + 2 * v) + 8 * hi;
        qf[c].u[v] = *reinterpret_cast<const u32*>(q + rowOff + kk);
      }
  }

  // phase 1: S[16][576] = QK^T * scale + bias
  for (int jt = wave; jt < NSEQ / 16; jt += 8) {
    int colBase = jt * 16;
    v8f acc;
    #pragma unroll
    for (int r = 0; r < 8; r++) acc[r] = 0.f;
    #pragma unroll
    for (int c = 0; c < 3; c++) {
      Frag bf;
      #pragma unroll
      for (int v = 0; v < 8; v++) {
        int kb = c * 32 + ((v < 4) ? 2 * v : 8 + 2 * v) + 8 * hi;  // contiguous pair along HD of K-row
        bf.u[v] = *reinterpret_cast<const u32*>(
            kx + (seqBase + colBase + m) * 768 + h * 96 + kb);
      }
      acc = __builtin_amdgcn_wmma_f32_16x16x32_bf16(
          false, qf[c].v, false, bf.v, (short)0, acc, false, false);
    }
    #pragma unroll
    for (int r = 0; r < 8; r++) {
      int mr = 8 * hi + r, col = colBase + m;
      S[mr * NSEQ + col] = acc[r] * scale +
                           biasNN[(size_t)(rowBase + mr) * NSEQ + col];
    }
  }
  __syncthreads();

  // phase 2: softmax per row (16 threads per row)
  {
    int row = tid >> 4, sub = tid & 15;
    float lm = -3.0e38f;
    for (int c = sub; c < NSEQ; c += 16) lm = fmaxf(lm, S[row * NSEQ + c]);
    red[tid] = lm;
    __syncthreads();
    for (int off = 8; off > 0; off >>= 1) {
      if (sub < off) red[tid] = fmaxf(red[tid], red[tid + off]);
      __syncthreads();
    }
    float mx = red[row * 16];
    __syncthreads();
    float ls = 0.f;
    for (int c = sub; c < NSEQ; c += 16) {
      float e = __expf(S[row * NSEQ + c] - mx);
      S[row * NSEQ + c] = e;
      ls += e;
    }
    red[tid] = ls;
    __syncthreads();
    for (int off = 8; off > 0; off >>= 1) {
      if (sub < off) red[tid] += red[tid + off];
      __syncthreads();
    }
    float inv = 1.0f / red[row * 16];
    __syncthreads();
    for (int c = sub; c < NSEQ; c += 16) S[row * NSEQ + c] *= inv;
  }
  __syncthreads();

  // phase 3: O = P V ; K=576 split in 32-wide chunks round-robin over waves
  {
    v8f acc[6];
    #pragma unroll
    for (int t = 0; t < 6; t++)
      #pragma unroll
      for (int r = 0; r < 8; r++) acc[t][r] = 0.f;
    for (int ch = wave; ch < NSEQ / 32; ch += 8) {
      Frag pf;
      #pragma unroll
      for (int v = 0; v < 8; v++) {
        int kk = ch * 32 + ((v < 4) ? 2 * v : 8 + 2 * v) + 8 * hi;
        u16 lo = f2bf(S[m * NSEQ + kk]);
        u16 hh = f2bf(S[m * NSEQ + kk + 1]);
        pf.u[v] = (u32)lo | ((u32)hh << 16);
      }
      #pragma unroll
      for (int t = 0; t < 6; t++) {
        Frag bf;
        #pragma unroll
        for (int v = 0; v < 8; v++) {
          int k0 = ch * 32 + ((v < 4) ? 2 * v : 8 + 2 * v) + 8 * hi;
          u16 lo = vx[(seqBase + k0) * 768 + h * 96 + t * 16 + m];
          u16 hh = vx[(seqBase + k0 + 1) * 768 + h * 96 + t * 16 + m];
          bf.u[v] = (u32)lo | ((u32)hh << 16);
        }
        acc[t] = __builtin_amdgcn_wmma_f32_16x16x32_bf16(
            false, pf.v, false, bf.v, (short)0, acc[t], false, false);
      }
    }
    #pragma unroll
    for (int t = 0; t < 6; t++)
      #pragma unroll
      for (int r = 0; r < 8; r++)
        atomicAdd(&O[(8 * hi + r) * 96 + t * 16 + m], acc[t][r]);  // ds_add_f32
  }
  __syncthreads();

  // phase 4: write bf16 output [B,N,H*HD]
  for (int i = tid; i < 16 * 96; i += 256) {
    int mr = i / 96, c = i % 96;
    outB[(seqBase + rowBase + mr) * 768 + h * 96 + c] = f2bf(O[i]);
  }
}

// ---------------------------------------------------------------------------
// Host launcher
// ---------------------------------------------------------------------------
extern "C" void kernel_launch(void* const* d_in, const int* in_sizes, int n_in,
                              void* d_out, int out_size, void* d_ws, size_t ws_size,
                              hipStream_t stream) {
  (void)in_sizes; (void)n_in; (void)out_size; (void)ws_size;
  const float* x_in = (const float*)d_in[0];
  const float* rowe = (const float*)d_in[1];
  const float* cole = (const float*)d_in[2];
  const float* posw = (const float*)d_in[3];
  const float* posb = (const float*)d_in[4];
  const float* relb = (const float*)d_in[5];
  const float* ln1g = (const float*)d_in[6];
  const float* ln1b = (const float*)d_in[7];
  const float* ln2g = (const float*)d_in[8];
  const float* ln2b = (const float*)d_in[9];
  const float* Wq = (const float*)d_in[10]; const float* bq = (const float*)d_in[11];
  const float* Wk = (const float*)d_in[12]; const float* bk = (const float*)d_in[13];
  const float* Wv = (const float*)d_in[14]; const float* bv = (const float*)d_in[15];
  const float* Wo = (const float*)d_in[16]; const float* bo = (const float*)d_in[17];
  const float* W1 = (const float*)d_in[18]; const float* b1 = (const float*)d_in[19];
  const float* W2 = (const float*)d_in[20]; const float* b2 = (const float*)d_in[21];

  const int M = 32 * 576, D = 768, MLPD = 3072, N = 576, L = 2;

  char* ws = (char*)d_ws;
  size_t off = 0;
  auto carve = [&](size_t bytes) -> char* {
    char* p = ws + off;
    off = (off + bytes + 255) & ~(size_t)255;
    return p;
  };
  float* xbuf  = (float*)carve((size_t)M * D * 4);
  float* pep   = (float*)carve((size_t)N * D * 4);
  float* biasW = (float*)carve((size_t)N * N * 4);
  u16* act = (u16*)carve((size_t)M * D * 2);
  u16* qb  = (u16*)carve((size_t)M * D * 2);
  u16* kb  = (u16*)carve((size_t)M * D * 2);
  u16* vb  = (u16*)carve((size_t)M * D * 2);
  u16* ao  = (u16*)carve((size_t)M * D * 2);
  u16* hb  = (u16*)carve((size_t)M * MLPD * 2);
  u16* wqB = (u16*)carve((size_t)L * D * D * 2);
  u16* wkB = (u16*)carve((size_t)L * D * D * 2);
  u16* wvB = (u16*)carve((size_t)L * D * D * 2);
  u16* woB = (u16*)carve((size_t)L * D * D * 2);
  u16* w1B = (u16*)carve((size_t)L * D * MLPD * 2);
  u16* w2B = (u16*)carve((size_t)L * MLPD * D * 2);

  auto cvt = [&](const float* s, u16* d, size_t n) {
    cvt_kernel<<<dim3((unsigned)((n + 255) / 256)), 256, 0, stream>>>(s, d, (int)n);
  };
  cvt(Wq, wqB, (size_t)L * D * D);
  cvt(Wk, wkB, (size_t)L * D * D);
  cvt(Wv, wvB, (size_t)L * D * D);
  cvt(Wo, woB, (size_t)L * D * D);
  cvt(W1, w1B, (size_t)L * D * MLPD);
  cvt(W2, w2B, (size_t)L * MLPD * D);

  peproj_kernel<<<N, 256, 0, stream>>>(rowe, cole, posw, posb, pep);
  addpe_kernel<<<(unsigned)((size_t)M * D / 256), 256, 0, stream>>>(x_in, pep, xbuf);
  bias_kernel<<<(N * N) / 256, 256, 0, stream>>>(relb, biasW);

  for (int l = 0; l < L; l++) {
    ln_kernel<<<M, 256, 0, stream>>>(xbuf, ln1g + l * D, ln1b + l * D, act);
    gemm_bf16_kernel<<<dim3(M / 128, D / 64), 256, 0, stream>>>(
        act, wqB + (size_t)l * D * D, bq + l * D, D, D, 0, nullptr, qb);
    gemm_bf16_kernel<<<dim3(M / 128, D / 64), 256, 0, stream>>>(
        act, wkB + (size_t)l * D * D, bk + l * D, D, D, 0, nullptr, kb);
    gemm_bf16_kernel<<<dim3(M / 128, D / 64), 256, 0, stream>>>(
        act, wvB + (size_t)l * D * D, bv + l * D, D, D, 0, nullptr, vb);
    attn_kernel<<<dim3(N / 16, 8, 32), 256, 0, stream>>>(qb, kb, vb, biasW, ao);
    gemm_bf16_kernel<<<dim3(M / 128, D / 64), 256, 0, stream>>>(
        ao, woB + (size_t)l * D * D, bo + l * D, D, D, 2, xbuf, nullptr);
    ln_kernel<<<M, 256, 0, stream>>>(xbuf, ln2g + l * D, ln2b + l * D, act);
    gemm_bf16_kernel<<<dim3(M / 128, MLPD / 64), 256, 0, stream>>>(
        act, w1B + (size_t)l * D * MLPD, b1 + l * MLPD, D, MLPD, 1, nullptr, hb);
    gemm_bf16_kernel<<<dim3(M / 128, D / 64), 256, 0, stream>>>(
        hb, w2B + (size_t)l * MLPD * D, b2 + l * D, MLPD, D, 2, xbuf, nullptr);
  }

  hipMemcpyAsync(d_out, xbuf, (size_t)M * D * 4, hipMemcpyDeviceToDevice, stream);
}
